// DiffPool_63118839382457
// MI455X (gfx1250) — matrix operand order repaired and checked by reading, not compiled, not run on hardware
//
#include <hip/hip_runtime.h>
#include <hip/hip_bf16.h>

// ---------------------------------------------------------------------------
// DiffPool forward for MI455X (gfx1250, wave32, WMMA + TDM).
// Bandwidth-bound: x = 256MB fp32 dominates (≈11us @ 23.3TB/s). GEMM (8.6
// GFLOP, K_OUT=64) uses v_wmma_f32_16x16x32_bf16; gumbel tile (32KB/block,
// no dtype conversion needed) is staged via the Tensor Data Mover
// (tensor_load_to_lds + s_wait_tensorcnt); softmax/argmax fused in-register
// via the ISA C-matrix layout + width-16 shuffles. Pooled scatter uses global
// f32 atomics (64MB target array is L2-resident; L2=192MB).
// ---------------------------------------------------------------------------

typedef __bf16 bf16_t;
typedef bf16_t v16bf __attribute__((ext_vector_type(16)));
typedef float  v8f   __attribute__((ext_vector_type(8)));
typedef unsigned int u32x4 __attribute__((ext_vector_type(4)));
typedef int          i32x4 __attribute__((ext_vector_type(4)));
typedef int          i32x8 __attribute__((ext_vector_type(8)));

constexpr int   D_IN    = 256;
constexpr int   K_OUT   = 64;
constexpr int   N_NODES = 262144;
constexpr int   N_EDGES = 4194304;
constexpr int   BATCH   = 1024;
constexpr int   ROWS    = 128;        // node rows per block (8 waves x 16)
constexpr int   XPB     = D_IN + 8;   // bf16 pitch: 264 elems = 528B -> +4 banks/row
constexpr int   WPB     = D_IN + 8;
constexpr float INV_TAU = 1.0f;

#ifndef __has_builtin
#define __has_builtin(x) 0
#endif
#if __has_builtin(__builtin_amdgcn_tensor_load_to_lds) && \
    __has_builtin(__builtin_amdgcn_s_wait_tensorcnt)
#define USE_TDM 1
#else
#define USE_TDM 0
#endif

// ---------------- kernel 0: zero the pooled-output region ------------------
__global__ void k_zero(float4* __restrict__ p, int n4) {
    int g = blockIdx.x * blockDim.x + threadIdx.x;
    if (g < n4) p[g] = make_float4(0.f, 0.f, 0.f, 0.f);
}

// ---------------- kernel 1: GEMM(WMMA) + gumbel softmax + argmax -----------
__global__ __launch_bounds__(256)
void k_assign(const float* __restrict__ x, const float* __restrict__ Wg,
              const float* __restrict__ bg, const float* __restrict__ gum,
              float* __restrict__ outAssign, int* __restrict__ idxArr)
{
    __shared__ bf16_t sX[ROWS * XPB];      // 67.6 KB
    __shared__ bf16_t sW[K_OUT * WPB];     // 33.8 KB
    __shared__ float  sGum[ROWS * K_OUT];  // 32.0 KB (TDM destination)
    __shared__ float  sB[K_OUT];

    const int tid  = threadIdx.x;
    const int base = blockIdx.x * ROWS;

#if USE_TDM
    // --- TDM: DMA the contiguous 32KB gumbel tile (128 rows x 64 f32) into
    // LDS. Issued once by wave 0 (TDM ignores EXEC); D# per ISA ch.8.
    if (tid < 32) {
        const unsigned long long ga =
            (unsigned long long)(const void*)(gum + (size_t)base * K_OUT);
        const unsigned ldsAddr = (unsigned)(unsigned long long)(const void*)sGum;
        constexpr unsigned NDW = (unsigned)(ROWS * K_OUT);   // 8192 dwords
        // Group0: count=1 | lds_addr | global_addr | type=2 ("image")
        u32x4 g0 = { 1u,
                     ldsAddr,
                     (unsigned)(ga & 0xffffffffull),
                     (unsigned)((ga >> 32) & 0x01ffffffull) | (2u << 30) };
        // Group1: data_size=4B; tensor_dim0 = tile_dim0 = NDW; tensor_dim1 = 1;
        // tile_dim1 = 1; stride0 = NDW.
        i32x8 g1 = { (int)(2u << 16),                      // data_size code 2
                     (int)((NDW & 0xffffu) << 16),         // tensor_dim0[15:0]
                     (int)(((NDW >> 16) & 0xffffu) | (1u << 16)), // td0 hi | td1 lo
                     (int)((1u << 16)),                    // td1 hi=0 | tile_dim0.. 
                     0, 0, 0, 0 };
        // fix tile_dim0 (bits[127:112] => dword3[31:16]) and tile_dim1
        // (bits[143:128] => dword4[15:0]):
        g1[3] = (int)((NDW & 0xffffu) << 16);              // tile_dim0 = NDW
        g1[4] = (int)1;                                    // tile_dim1 = 1
        g1[5] = (int)NDW;                                  // tensor_dim0_stride lo
        g1[6] = 0;                                         // stride0 hi | stride1 lo
        g1[7] = 0;                                         // stride1 hi
        i32x4 gz = { 0, 0, 0, 0 };                         // groups 2/3: <=2D
#if __clang_major__ >= 23
        i32x8 gz8 = { 0, 0, 0, 0, 0, 0, 0, 0 };
        __builtin_amdgcn_tensor_load_to_lds(g0, g1, gz, gz, gz8, 0);
#else
        __builtin_amdgcn_tensor_load_to_lds(g0, g1, gz, gz, 0);
#endif
        __builtin_amdgcn_s_wait_tensorcnt(0);
    }
#else
    for (int i = tid; i < ROWS * K_OUT / 4; i += 256) {
        reinterpret_cast<float4*>(sGum)[i] =
            reinterpret_cast<const float4*>(gum + (size_t)base * K_OUT)[i];
    }
#endif

    // stage W: f32 -> bf16, row-major [K_OUT][D_IN] with padded pitch
    for (int i = tid; i < K_OUT * (D_IN / 4); i += 256) {
        int r = i >> 6;                // D_IN/4 = 64 float4 chunks per row
        int c = (i & 63) << 2;
        float4 v = reinterpret_cast<const float4*>(Wg)[i];
        bf16_t* p = &sW[r * WPB + c];
        p[0] = (bf16_t)v.x; p[1] = (bf16_t)v.y;
        p[2] = (bf16_t)v.z; p[3] = (bf16_t)v.w;
    }
    // stage x tile: 128 rows x 256 f32 -> bf16 (coalesced b128 loads)
    for (int i = tid; i < ROWS * (D_IN / 4); i += 256) {
        int r = i >> 6, cc = i & 63;
        float4 v = reinterpret_cast<const float4*>(x)[(size_t)(base + r) * (D_IN / 4) + cc];
        bf16_t* p = &sX[r * XPB + (cc << 2)];
        p[0] = (bf16_t)v.x; p[1] = (bf16_t)v.y;
        p[2] = (bf16_t)v.z; p[3] = (bf16_t)v.w;
    }
    if (tid < K_OUT) sB[tid] = bg[tid];
    __syncthreads();

    const int wave = tid >> 5;
    const int lane = tid & 31;
    const int l15  = lane & 15;
    const int hi   = lane >> 4;
    const int arow = ((wave << 4) + l15) * XPB;   // A-matrix row for this lane

    v8f acc[4];
#pragma unroll
    for (int nt = 0; nt < 4; ++nt)
#pragma unroll
        for (int j = 0; j < 8; ++j) acc[nt][j] = 0.0f;

    // K loop: 8 chunks of 32 (D_IN = 256)
    for (int kc = 0; kc < D_IN / 32; ++kc) {
        // A fragment: 16x32 bf16: VGPR j -> K=2j,2j+1 ;
        // lanes 16-31 shifted by +16 in K (matches B's contiguous order).
        union { v16bf v; unsigned u[8]; } A;
        const int kbase = kc * 32;
#pragma unroll
        for (int j = 0; j < 8; ++j) {
            int kp = (j << 1) + (hi << 4);
            A.u[j] = *reinterpret_cast<const unsigned*>(&sX[arow + kbase + kp]);
        }
#pragma unroll
        for (int nt = 0; nt < 4; ++nt) {
            // B fragment: 32x16 bf16; lane holds column (nt*16+l15), 16
            // consecutive K starting at kc*32 + hi*16 -> two ds_load_b128.
            union { v16bf v; uint4 q[2]; } B;
            const int col = (nt << 4) + l15;
            const uint4* bp = reinterpret_cast<const uint4*>(
                &sW[col * WPB + kbase + (hi << 4)]);
            B.q[0] = bp[0]; B.q[1] = bp[1];
            acc[nt] = __builtin_amdgcn_wmma_f32_16x16x32_bf16(
                false, A.v, false, B.v, (short)0, acc[nt], false, false);
        }
    }

    // bias (per output column)
#pragma unroll
    for (int nt = 0; nt < 4; ++nt) {
        float bb = sB[(nt << 4) + l15];
#pragma unroll
        for (int j = 0; j < 8; ++j) acc[nt][j] += bb;
    }

    // C layout: element acc[nt][j] = logits[row = j + 8*hi][col = nt*16 + l15]
    // Per row: 64 scores live in 16 lanes (one hi-half) x 4 regs -> width-16
    // shuffle reductions for max/argmax and the exp-sum.
#pragma unroll
    for (int j = 0; j < 8; ++j) {
        const int lr = (wave << 4) + j + (hi << 3);
        const int gr = base + lr;
        float s[4];
#pragma unroll
        for (int nt = 0; nt < 4; ++nt)
            s[nt] = (acc[nt][j] + sGum[lr * K_OUT + (nt << 4) + l15]) * INV_TAU;

        float mx = s[0]; int ac = l15;
#pragma unroll
        for (int nt = 1; nt < 4; ++nt) {
            int c = (nt << 4) + l15;
            if (s[nt] > mx) { mx = s[nt]; ac = c; }
        }
#pragma unroll
        for (int d = 1; d < 16; d <<= 1) {
            float om = __shfl_xor(mx, d, 16);
            int   oc = __shfl_xor(ac, d, 16);
            if (om > mx || (om == mx && oc < ac)) { mx = om; ac = oc; }
        }
        float e[4], ls = 0.0f;
#pragma unroll
        for (int nt = 0; nt < 4; ++nt) { e[nt] = expf(s[nt] - mx); ls += e[nt]; }
#pragma unroll
        for (int d = 1; d < 16; d <<= 1) ls += __shfl_xor(ls, d, 16);
        const float inv = 1.0f / ls;
#pragma unroll
        for (int nt = 0; nt < 4; ++nt) {
            int c = (nt << 4) + l15;
            float ys = e[nt] * inv;
            float h  = (c == ac) ? 1.0f : 0.0f;
            // match reference fp order: (y_hard + y_soft) - y_soft
            outAssign[(size_t)gr * K_OUT + c] = (h + ys) - ys;
        }
        if (l15 == 0) idxArr[gr] = ac;
    }
}

// ---------------- kernel 2: edge relabel (gather idx, emit as f32) ---------
__global__ void k_edges(const int4* __restrict__ el, const int* __restrict__ idxArr,
                        float4* __restrict__ out, int n4)
{
    int g = blockIdx.x * blockDim.x + threadIdx.x;
    if (g >= n4) return;
    int4 e = el[g];
    out[g] = make_float4((float)idxArr[e.x], (float)idxArr[e.y],
                         (float)idxArr[e.z], (float)idxArr[e.w]);
}

// ---------------- kernel 3: pooled segment-sum via L2 atomics --------------
__global__ __launch_bounds__(256)
void k_pool(const float* __restrict__ x, const int* __restrict__ idxArr,
            const int* __restrict__ n2g, float* __restrict__ outP)
{
    const int n = blockIdx.x;            // one node per block, 256 = D_IN lanes
    const int t = threadIdx.x;
    const int seg = n2g[n] * K_OUT + idxArr[n];
    atomicAdd(&outP[(size_t)seg * D_IN + t], x[(size_t)n * D_IN + t]);
}

// ---------------------------------------------------------------------------
extern "C" void kernel_launch(void* const* d_in, const int* in_sizes, int n_in,
                              void* d_out, int out_size, void* d_ws, size_t ws_size,
                              hipStream_t stream)
{
    (void)in_sizes; (void)n_in; (void)out_size; (void)ws_size;
    const float* x   = (const float*)d_in[0];
    const float* W   = (const float*)d_in[1];
    const float* b   = (const float*)d_in[2];
    const float* gum = (const float*)d_in[3];
    const int*   el  = (const int*)d_in[4];
    const int*   n2g = (const int*)d_in[5];
    // d_in[6] = batch_size scalar (device); layout is compile-time constant.

    float* outE = (float*)d_out;                                 // [N_EDGES*2]
    float* outP = outE + (size_t)N_EDGES * 2;                    // [BATCH*K_OUT*D_IN]
    float* outA = outP + (size_t)BATCH * K_OUT * D_IN;           // [N_NODES*K_OUT]
    int*   idxA = (int*)d_ws;                                    // [N_NODES]

    // 0) zero pooled region (harness poisons d_out; atomics need zeros)
    {
        int n4 = (BATCH * K_OUT * D_IN) / 4;
        k_zero<<<(n4 + 255) / 256, 256, 0, stream>>>((float4*)outP, n4);
    }
    // 1) WMMA logits + gumbel softmax + argmax + assignment
    k_assign<<<N_NODES / ROWS, 256, 0, stream>>>(x, W, b, gum, outA, idxA);
    // 2) edge relabel
    {
        int n4 = (N_EDGES * 2) / 4;
        k_edges<<<(n4 + 255) / 256, 256, 0, stream>>>((const int4*)el, idxA,
                                                      (float4*)outE, n4);
    }
    // 3) pooled segment sum
    k_pool<<<N_NODES, 256, 0, stream>>>(x, idxA, n2g, outP);
}
